// FCCaps_28363964022836
// MI455X (gfx1250) — compile-verified, matrix operand order checked by reading, not run on hardware
//
#include <hip/hip_runtime.h>
#include <hip/hip_bf16.h>
#include <cstdint>
#include <cstddef>

// FCCaps KDE-routing capsule layer for gfx1250 (MI455X).
//   u_hat = einsum('ijkl,bil->bijk', W, x)  -> bf16-split WMMA (v_wmma_f32_16x16x32_bf16)
//   3 fixed routing iterations, fused so u_hat is streamed once per iteration.
// Workspace requirement: ~148 MB in d_ws.

typedef __attribute__((ext_vector_type(16))) __bf16 v16bf;
typedef __attribute__((ext_vector_type(8)))  float  v8f;
typedef __attribute__((ext_vector_type(4)))  float  f4;

#define B_   32
#define NIN  1024
#define NOUT 64
#define DIN  128
#define DOUT 16
#define EPSF 1e-7f

__device__ __forceinline__ v8f wmma_bf16(v16bf a, v16bf b, v8f c) {
  // 8 args: (neg_a, A, neg_b, B, c_mod, C, reuse_a, reuse_b)
  return __builtin_amdgcn_wmma_f32_16x16x32_bf16(false, a, false, b, (short)0, c, false, false);
}

// Split 8 contiguous fp32 into bf16 hi/lo halves, filling vector lanes [base, base+8).
__device__ __forceinline__ void load8_split(const float* __restrict__ p,
                                            v16bf& hi, v16bf& lo, int base) {
#pragma unroll
  for (int t = 0; t < 8; ++t) {
    float f   = p[t];
    __bf16 h  = (__bf16)f;
    hi[base + t] = h;
    lo[base + t] = (__bf16)(f - (float)h);
  }
}

// ---------------------------------------------------------------------------
// Kernel 1: u_hat via WMMA. grid = (NIN, 8), block = 256 (8 waves).
// Wave w of block (i, by) computes M-tile j = by*8 + w (16 rows = one output
// capsule) against both batch halves (two 16-wide N-tiles).
// A layout (16-bit, 16x32): lanes 0-15 -> M=lane, K {0..7} in v0..3 and
// {16..23} in v4..7; lanes 16-31 -> K {8..15} / {24..31}. B mirrors with
// lane = n (column-major columns of x are contiguous in memory).
// ---------------------------------------------------------------------------
__global__ __launch_bounds__(256) void caps_uhat(const float* __restrict__ W,
                                                 const float* __restrict__ X,
                                                 float* __restrict__ U) {
  const int i    = blockIdx.x;
  const int wave = threadIdx.x >> 5;
  const int lane = threadIdx.x & 31;
  const int j    = blockIdx.y * 8 + wave;    // output capsule / M-tile
  const int half = lane >> 4;
  const int row  = lane & 15;

  const float* Wr = W + ((size_t)i * NOUT * DOUT + (size_t)j * DOUT + row) * DIN;
  const float* X0 = X + ((size_t)row        * NIN + i) * DIN;   // batch row
  const float* X1 = X + ((size_t)(row + 16) * NIN + i) * DIN;   // batch row+16

  v8f c0 = {};
  v8f c1 = {};
#pragma unroll
  for (int kc = 0; kc < 4; ++kc) {           // K = 128 in 4 chunks of 32
    const int s0 = kc * 32 + half * 8;
    const int s1 = s0 + 16;
    v16bf ah, al, b0h, b0l, b1h, b1l;
    load8_split(Wr + s0, ah, al, 0);
    load8_split(Wr + s1, ah, al, 8);
    load8_split(X0 + s0, b0h, b0l, 0);
    load8_split(X0 + s1, b0h, b0l, 8);
    load8_split(X1 + s0, b1h, b1l, 0);
    load8_split(X1 + s1, b1h, b1l, 8);
    // fp32 ~= hi*hi + hi*lo + lo*hi (bf16 split, fp32 accumulate)
    c0 = wmma_bf16(ah, b0h, c0);
    c0 = wmma_bf16(ah, b0l, c0);
    c0 = wmma_bf16(al, b0h, c0);
    c1 = wmma_bf16(ah, b1h, c1);
    c1 = wmma_bf16(ah, b1l, c1);
    c1 = wmma_bf16(al, b1h, c1);
  }

  // D layout: n = lane&15, m = (lane>>4)*8 + r  -> k = m (tile is one j).
  const int kb = half * 8;
  float* U0 = U + (((size_t)row        * NIN + i) * NOUT + j) * DOUT + kb;
  float* U1 = U + (((size_t)(row + 16) * NIN + i) * NOUT + j) * DOUT + kb;
  *(f4*)(U0)     = (f4){c0[0], c0[1], c0[2], c0[3]};
  *(f4*)(U0 + 4) = (f4){c0[4], c0[5], c0[6], c0[7]};
  *(f4*)(U1)     = (f4){c1[0], c1[1], c1[2], c1[3]};
  *(f4*)(U1 + 4) = (f4){c1[4], c1[5], c1[6], c1[7]};
}

// ---------------------------------------------------------------------------
// Kernel 2: one routing iteration. grid = (32 i-chunks, B), block = 256.
// Thread (g = tid>>6, j = tid&63) loops 8 input capsules; per (b,i,j) it
// reads u (16 floats), applies the deferred logit update dd(v_prev), does a
// 64-wide softmax, and accumulates c and c*u into thread-exclusive LDS.
// Block partials go to disjoint global slices (deterministic reduce later).
// ---------------------------------------------------------------------------
__global__ __launch_bounds__(256) void caps_route(const float* __restrict__ U,
                                                  const float* __restrict__ bparam,
                                                  float* __restrict__ logits,
                                                  const float* __restrict__ v_in,
                                                  const float* __restrict__ vn_in,
                                                  float* __restrict__ vp_part,
                                                  float* __restrict__ s_part,
                                                  int iter) {
  __shared__ float vs[NOUT][DOUT];
  __shared__ float vns[NOUT];
  __shared__ float vps[4][NOUT][DOUT];
  __shared__ float Ss[4][NOUT];
  __shared__ float redm[4][2];
  __shared__ float reds[4][2];

  const int b   = blockIdx.y;
  const int tid = threadIdx.x;
  const int g   = tid >> 6;
  const int j   = tid & 63;
  const int lig = j & 31;        // lane within the group's wave
  const int wig = j >> 5;        // which of the group's two waves

#pragma unroll
  for (int k = 0; k < DOUT; ++k) vps[g][j][k] = 0.0f;
  Ss[g][j] = 0.0f;
  if (iter > 0) {
    for (int t = tid; t < NOUT * DOUT; t += 256)
      vs[t >> 4][t & 15] = v_in[(size_t)b * NOUT * DOUT + t];
    if (tid < NOUT) vns[tid] = vn_in[b * NOUT + tid];
  }
  __syncthreads();

  const int ibase = blockIdx.x * 32;
  for (int step = 0; step < 8; ++step) {
    const int i = ibase + g * 8 + step;
    const float* up = U + (((size_t)b * NIN + i) * NOUT + j) * DOUT;
    float u[DOUT];
    {
      const f4* u4 = (const f4*)up;
      f4 a0 = u4[0], a1 = u4[1], a2 = u4[2], a3 = u4[3];
#pragma unroll
      for (int t = 0; t < 4; ++t) {
        u[t] = a0[t]; u[4 + t] = a1[t]; u[8 + t] = a2[t]; u[12 + t] = a3[t];
      }
    }
    float s = 0.0f;
#pragma unroll
    for (int k = 0; k < DOUT; ++k) s += u[k] * u[k];

    const size_t lidx = ((size_t)b * NIN + i) * NOUT + j;
    float lg;
    if (iter == 0) {
      lg = bparam[i * NOUT + j];
    } else {
      float duv = 0.0f;
#pragma unroll
      for (int k = 0; k < DOUT; ++k) duv += u[k] * vs[j][k];
      const float inv  = rsqrtf(s + EPSF);
      const float gq   = s / (1.0f + s);
      const float gs   = gq * inv;              // u_hat_sq = gs * u
      const float uhs2 = gs * gs * s;           // ||u_hat_sq||^2
      const float dd   = 1.0f - uhs2 - vns[j] + 2.0f * gs * duv;
      const float base = (iter == 1) ? bparam[i * NOUT + j] : logits[lidx];
      lg = base + dd;
      if (iter == 1) logits[lidx] = lg;         // needed by iter 2
    }

    // softmax over the 64 output capsules (2 waves per group)
    float m = lg;
#pragma unroll
    for (int off = 16; off >= 1; off >>= 1) m = fmaxf(m, __shfl_xor(m, off));
    if (lig == 0) redm[g][wig] = m;
    __syncthreads();
    m = fmaxf(redm[g][0], redm[g][1]);
    float e = __expf(lg - m);
    float se = e;
#pragma unroll
    for (int off = 16; off >= 1; off >>= 1) se += __shfl_xor(se, off);
    if (lig == 0) reds[g][wig] = se;
    __syncthreads();
    se = reds[g][0] + reds[g][1];
    const float c = e / se;                     // renorm over i deferred to reduce

    Ss[g][j] += c;
#pragma unroll
    for (int k = 0; k < DOUT; ++k) vps[g][j][k] += c * u[k];
    __syncthreads();                            // protect red arrays for next step
  }

  // flush block partials (each block owns a disjoint slice -> deterministic)
  {
    const int jj = tid & 63;
    const int kq = (tid >> 6) * 4;
    float* vpo = vp_part + (((size_t)blockIdx.x * B_ + b) * NOUT + jj) * DOUT;
#pragma unroll
    for (int k = kq; k < kq + 4; ++k)
      vpo[k] = vps[0][jj][k] + vps[1][jj][k] + vps[2][jj][k] + vps[3][jj][k];
    if (tid < NOUT)
      s_part[((size_t)blockIdx.x * B_ + b) * NOUT + tid] =
          Ss[0][tid] + Ss[1][tid] + Ss[2][tid] + Ss[3][tid];
  }
}

// ---------------------------------------------------------------------------
// Kernel 3: reduce partials, renormalize over i, squash -> v / ||v||^2.
// grid = 128, block = 256; thread t handles (entry e = t>>4, k = t&15).
// ---------------------------------------------------------------------------
__global__ __launch_bounds__(256) void caps_squash(const float* __restrict__ vp_part,
                                                   const float* __restrict__ s_part,
                                                   float* __restrict__ v_out,
                                                   float* __restrict__ vn_out,
                                                   float* __restrict__ act_out,
                                                   int final_iter) {
  const int t = blockIdx.x * 256 + threadIdx.x;   // 0 .. B*NOUT*DOUT-1
  const int e = t >> 4;
  const int k = t & 15;

  float vp = 0.0f, S = 0.0f;
#pragma unroll 4
  for (int c = 0; c < 32; ++c) {
    vp += vp_part[((size_t)c * (B_ * NOUT) + e) * DOUT + k];
    S  += s_part[(size_t)c * (B_ * NOUT) + e];
  }
  const float x = vp / S;                          // v_pre
  float sq = x * x;
#pragma unroll
  for (int off = 8; off >= 1; off >>= 1) sq += __shfl_xor(sq, off, 16);
  const float gq  = sq / (1.0f + sq);
  const float inv = rsqrtf(sq + EPSF);
  const float gv  = gq * inv;
  v_out[t] = gv * x;                               // poses when final
  if (k == 0) {
    const float vn2 = gv * gv * sq;                // ||v||^2
    if (final_iter) act_out[e] = sqrtf(vn2);
    else            vn_out[e]  = vn2;
  }
}

// ---------------------------------------------------------------------------
extern "C" void kernel_launch(void* const* d_in, const int* in_sizes, int n_in,
                              void* d_out, int out_size, void* d_ws, size_t ws_size,
                              hipStream_t stream) {
  const float* x  = (const float*)d_in[0];   // (B, NIN, DIN)
  const float* W  = (const float*)d_in[1];   // (NIN, NOUT, DOUT, DIN)
  const float* bp = (const float*)d_in[2];   // (NIN, NOUT, 1)
  float* out = (float*)d_out;                // poses (B,NOUT,DOUT) ++ activations (B,NOUT)

  float* ws = (float*)d_ws;
  const size_t U_elems  = (size_t)B_ * NIN * NOUT * DOUT;   // 33,554,432
  const size_t L_elems  = (size_t)B_ * NIN * NOUT;          //  2,097,152
  const size_t VP_elems = (size_t)32 * B_ * NOUT * DOUT;    //  1,048,576
  const size_t SP_elems = (size_t)32 * B_ * NOUT;           //     65,536
  float* U       = ws;
  float* logits  = U + U_elems;
  float* vp_part = logits + L_elems;
  float* s_part  = vp_part + VP_elems;
  float* v_buf   = s_part + SP_elems;                       // 32,768
  float* vn_buf  = v_buf + (size_t)B_ * NOUT * DOUT;        //  2,048

  // 1) u_hat via bf16-split WMMA
  caps_uhat<<<dim3(NIN, 8), 256, 0, stream>>>(W, x, U);

  // 2) three fused routing iterations
  for (int it = 0; it < 3; ++it) {
    caps_route<<<dim3(32, B_), 256, 0, stream>>>(U, bp, logits, v_buf, vn_buf,
                                                 vp_part, s_part, it);
    const int fin = (it == 2);
    caps_squash<<<128, 256, 0, stream>>>(vp_part, s_part,
                                         fin ? out : v_buf, vn_buf,
                                         out + (size_t)B_ * NOUT * DOUT, fin);
  }
}
